// Local_Att_19198503813519
// MI455X (gfx1250) — matrix operand order repaired and checked by reading, not compiled, not run on hardware
//
#include <hip/hip_runtime.h>
#include <hip/hip_bf16.h>

#define Bc   16
#define Sc   2048
#define HD   512
#define AD   128
#define WINc 5
#define Tc   (Sc - WINc + 1)   // 2044
#define LN_EPS 1e-5f

typedef __attribute__((ext_vector_type(16))) __bf16 v16bf;
typedef __attribute__((ext_vector_type(8)))  short  v8s;
typedef __attribute__((ext_vector_type(8)))  float  v8f;

// fp32 -> bf16 bits, round-to-nearest-even
__device__ inline short f2bf(float f) {
    union { float f; unsigned u; } c; c.f = f;
    unsigned u = c.u;
    u += 0x7FFFu + ((u >> 16) & 1u);
    return (short)(u >> 16);
}

__device__ inline v16bf combine(v8s lo, v8s hi) {
    union { struct { v8s lo, hi; } p; v16bf b; } u;
    u.p.lo = lo; u.p.hi = hi;
    return u.b;
}

__device__ inline v8f wmma_bf16(v16bf a, v16bf b, v8f c) {
    return __builtin_amdgcn_wmma_f32_16x16x32_bf16(false, a, false, b, (short)0, c, false, false);
}

// A fragment (16x32 bf16) from bf16 rows (LDS or global): two 128-bit loads.
// ISA 7.12.2: lane L row M=L&15; half=L>>4; e<8 -> K=kbase+half*8+e ; e>=8 -> +16
__device__ inline v16bf fragA(const short* base, int rowstride, int lane, int kbase) {
    const short* row = base + (long)(lane & 15) * rowstride + kbase + (lane >> 4) * 8;
    return combine(*(const v8s*)row, *(const v8s*)(row + 16));
}

// B fragment from pre-swizzled weights (global or LDS): 32 contiguous bytes/lane.
// frag_idx = nt*(K/32) + kc
__device__ inline v16bf fragB(const short* Wp, long frag_idx, int lane) {
    const short* p = Wp + (frag_idx * 32 + lane) * 16;
    return combine(*(const v8s*)p, *(const v8s*)(p + 8));
}

// ---------------------------------------------------------------------------
// Prep kernel 1: elementwise fp32 -> bf16 (for xs), float4 in / short4 out
// ---------------------------------------------------------------------------
__global__ void __launch_bounds__(256) cvt_bf_kernel(const float* __restrict__ in,
                                                     short* __restrict__ out, long nel) {
    long i = ((long)blockIdx.x * blockDim.x + threadIdx.x) * 4;
    if (i + 3 < nel) {
        float4 v = *(const float4*)(in + i);
        short4 o;
        o.x = f2bf(v.x); o.y = f2bf(v.y); o.z = f2bf(v.z); o.w = f2bf(v.w);
        *(short4*)(out + i) = o;
    } else {
        for (; i < nel; ++i) out[i] = f2bf(in[i]);
    }
}

// ---------------------------------------------------------------------------
// Prep kernel 2: pack fp32 weight [K][N] into WMMA B-fragment lane order
// ---------------------------------------------------------------------------
__global__ void __launch_bounds__(256) pack_w_kernel(const float* __restrict__ W,
                                                     short* __restrict__ out, int K, int N) {
    long total = (long)K * N;
    long i = (long)blockIdx.x * blockDim.x + threadIdx.x;
    if (i >= total) return;
    int e  = (int)(i & 15);
    int L  = (int)((i >> 4) & 31);
    long f = i >> 9;
    int KC = K >> 5;
    int nt = (int)(f / KC);
    int kc = (int)(f % KC);
    int k   = kc * 32 + (L >> 4) * 8 + (e & 7) + (e >> 3) * 16;
    int col = nt * 16 + (L & 15);
    out[i] = f2bf(W[(long)k * N + col]);
}

// ---------------------------------------------------------------------------
// Kernel A: keys = X@Wk + bk ; values = X@Wv + bv   (parallel, bandwidth-bound)
// 256 threads (8 waves); 5 N-tiles per wave; software-pipelined k-loop.
// ---------------------------------------------------------------------------
__global__ void __launch_bounds__(256) kv_kernel(const short* __restrict__ xsb,
                                                 const short* __restrict__ Wkp, const float* __restrict__ bk,
                                                 const short* __restrict__ Wvp, const float* __restrict__ bv,
                                                 float* __restrict__ keys, float* __restrict__ values) {
    const int tid = threadIdx.x, lane = tid & 31, wave = tid >> 5;
    const long rbase = (long)blockIdx.x * 16;
    const short* arow = xsb + rbase * HD;

    v8f acc[5];
#pragma unroll
    for (int j = 0; j < 5; ++j)
#pragma unroll
        for (int r = 0; r < 8; ++r) acc[j][r] = 0.f;

    // prologue: fragments for kc = 0
    v16bf a = fragA(arow, HD, lane, 0);
    v16bf b[5];
#pragma unroll
    for (int j = 0; j < 5; ++j) {
        int nt = wave + j * 8;
        b[j] = (nt < 8) ? fragB(Wkp, (long)nt * 16, lane)
                        : fragB(Wvp, (long)(nt - 8) * 16, lane);
    }

    for (int kc = 0; kc < 16; ++kc) {
        v16bf an, bn[5];
        if (kc < 15) {      // preload kc+1 while kc computes
            an = fragA(arow, HD, lane, (kc + 1) * 32);
#pragma unroll
            for (int j = 0; j < 5; ++j) {
                int nt = wave + j * 8;
                bn[j] = (nt < 8) ? fragB(Wkp, (long)nt * 16 + kc + 1, lane)
                                 : fragB(Wvp, (long)(nt - 8) * 16 + kc + 1, lane);
            }
        }
#pragma unroll
        for (int j = 0; j < 5; ++j) acc[j] = wmma_bf16(a, b[j], acc[j]);
        if (kc < 15) {
            a = an;
#pragma unroll
            for (int j = 0; j < 5; ++j) b[j] = bn[j];
        }
    }

    const int n = lane & 15, half = lane >> 4;
#pragma unroll
    for (int j = 0; j < 5; ++j) {
        int nt = wave + j * 8;
        if (nt < 8) {
            int col = nt * 16 + n;
            float bias = bk[col];
#pragma unroll
            for (int r = 0; r < 8; ++r)
                keys[(rbase + half * 8 + r) * AD + col] = acc[j][r] + bias;
        } else {
            int col = (nt - 8) * 16 + n;
            float bias = bv[col];
#pragma unroll
            for (int r = 0; r < 8; ++r)
                values[(rbase + half * 8 + r) * HD + col] = acc[j][r] + bias;
        }
    }
}

// ---------------------------------------------------------------------------
// Kernel B: persistent single-workgroup scan over T=2044 dependent steps.
// 512 threads = 16 waves. LDS: state buffer (aliased ctx/context), q, out,
// and the full packed Wq (128 KB) cached once. Total LDS = 184 KB (< 320 KB).
// ---------------------------------------------------------------------------
__global__ void __launch_bounds__(512) scan_kernel(const float* __restrict__ xs,
                                                   const short* __restrict__ Wqp, const float* __restrict__ bq,
                                                   const float* __restrict__ keys, const float* __restrict__ values,
                                                   const short* __restrict__ Wop, const float* __restrict__ bo,
                                                   const float* __restrict__ ln_g, const float* __restrict__ ln_b,
                                                   float* __restrict__ out) {
    __shared__ short abuf[Bc * HD];       // phase1: ctx (A) ; phase3: context (A) ; phase4: new ctx
    __shared__ float q_lds[Bc * AD];
    __shared__ float out_lds[Bc * HD];
    __shared__ short wq_lds[HD * AD];     // packed Wq, resident all 2044 steps (128 KB)

    const int tid = threadIdx.x, lane = tid & 31, wave = tid >> 5;
    const int n = lane & 15, half = lane >> 4;

    for (int i = tid; i < Bc * HD; i += 512) abuf[i] = 0;
    for (int i = tid * 8; i < HD * AD; i += 512 * 8)          // one-time Wq -> LDS
        *(v8s*)(wq_lds + i) = *(const v8s*)(Wqp + i);
    __syncthreads();

    for (int t = 0; t < Tc; ++t) {
        // ---- Phase 1: q = ctx@Wq + bq : 8 N-tiles on waves 0..7 (all-LDS) ----
        if (wave < 8) {
            v8f acc;
#pragma unroll
            for (int r = 0; r < 8; ++r) acc[r] = 0.f;
            for (int kc = 0; kc < 16; ++kc) {
                v16bf a = fragA(abuf, HD, lane, kc * 32);
                v16bf b = fragB(wq_lds, (long)wave * 16 + kc, lane);
                acc = wmma_bf16(a, b, acc);
            }
            int col = wave * 16 + n;
            float bias = bq[col];
#pragma unroll
            for (int r = 0; r < 8; ++r)
                q_lds[(half * 8 + r) * AD + col] = acc[r] + bias;
        } else {
            // warm the incoming K/V window rows for two batches per wave
            int b0 = (wave - 8) * 2;
            __builtin_prefetch(keys + ((long)b0 * Sc + t + WINc - 1) * AD + lane * 4, 0, 1);
            __builtin_prefetch(keys + ((long)(b0 + 1) * Sc + t + WINc - 1) * AD + lane * 4, 0, 1);
            __builtin_prefetch(values + ((long)b0 * Sc + t + WINc - 1) * HD + lane * 16, 0, 1);
            __builtin_prefetch(values + ((long)(b0 + 1) * Sc + t + WINc - 1) * HD + lane * 16, 0, 1);
        }
        __syncthreads();

        // ---- Phase 2: additive score + softmax + context ; wave b = batch b ----
        {
            const int b = wave;
            const int d0 = lane * 4;                       // AD = 32 lanes * 4
            float4 qv = *(const float4*)(q_lds + b * AD + d0);
            float sc[WINc], att[WINc];
            float mx = -1e30f;
#pragma unroll
            for (int w = 0; w < WINc; ++w) {
                float4 kv = *(const float4*)(keys + ((long)b * Sc + t + w) * AD + d0);
                float p = tanhf(qv.x + kv.x) + tanhf(qv.y + kv.y)
                        + tanhf(qv.z + kv.z) + tanhf(qv.w + kv.w);
#pragma unroll
                for (int off = 16; off > 0; off >>= 1) p += __shfl_xor(p, off, 32);
                sc[w] = p;
                mx = fmaxf(mx, p);
            }
            float den = 0.f;
#pragma unroll
            for (int w = 0; w < WINc; ++w) { att[w] = __expf(sc[w] - mx); den += att[w]; }
            float rden = 1.f / den;
#pragma unroll
            for (int w = 0; w < WINc; ++w) att[w] *= rden;

#pragma unroll
            for (int c0 = lane * 4; c0 < HD; c0 += 128) {
                float4 cv = make_float4(0.f, 0.f, 0.f, 0.f);
#pragma unroll
                for (int w = 0; w < WINc; ++w) {
                    float4 vv = *(const float4*)(values + ((long)b * Sc + t + w) * HD + c0);
                    cv.x += att[w] * vv.x; cv.y += att[w] * vv.y;
                    cv.z += att[w] * vv.z; cv.w += att[w] * vv.w;
                }
                short4 o;
                o.x = f2bf(cv.x); o.y = f2bf(cv.y); o.z = f2bf(cv.z); o.w = f2bf(cv.w);
                *(short4*)(abuf + b * HD + c0) = o;
            }
        }
        __syncthreads();

        // ---- Phase 3: out = context@Wo + bo : 32 N-tiles, 2 per wave,
        //      kc-outer / nt-inner, double-buffered B ----
        {
            const long f0 = (long)(wave * 2 + 0) * 16;
            const long f1 = (long)(wave * 2 + 1) * 16;
            v8f acc0, acc1;
#pragma unroll
            for (int r = 0; r < 8; ++r) { acc0[r] = 0.f; acc1[r] = 0.f; }

            v16bf a  = fragA(abuf, HD, lane, 0);
            v16bf b0 = fragB(Wop, f0, lane);
            v16bf b1 = fragB(Wop, f1, lane);
            for (int kc = 0; kc < 16; ++kc) {
                v16bf an, b0n, b1n;
                if (kc < 15) {
                    an  = fragA(abuf, HD, lane, (kc + 1) * 32);
                    b0n = fragB(Wop, f0 + kc + 1, lane);
                    b1n = fragB(Wop, f1 + kc + 1, lane);
                }
                acc0 = wmma_bf16(a, b0, acc0);
                acc1 = wmma_bf16(a, b1, acc1);
                if (kc < 15) { a = an; b0 = b0n; b1 = b1n; }
            }
            int col0 = (wave * 2 + 0) * 16 + n;
            int col1 = (wave * 2 + 1) * 16 + n;
            float bias0 = bo[col0], bias1 = bo[col1];
#pragma unroll
            for (int r = 0; r < 8; ++r) {
                out_lds[(half * 8 + r) * HD + col0] = acc0[r] + bias0;
                out_lds[(half * 8 + r) * HD + col1] = acc1[r] + bias1;
            }
        }
        __syncthreads();

        // ---- Phase 4: LayerNorm(out + residual) -> ys[t] and next ctx ----
        {
            const int b = wave;
            const float* res = xs + ((long)b * Sc + (t + 1 + WINc / 2)) * HD;
            float s1 = 0.f, s2 = 0.f;
#pragma unroll
            for (int c0 = lane * 4; c0 < HD; c0 += 128) {
                float4 o4 = *(const float4*)(out_lds + b * HD + c0);
                float4 r4 = *(const float4*)(res + c0);
                float x0 = o4.x + r4.x, x1 = o4.y + r4.y, x2 = o4.z + r4.z, x3 = o4.w + r4.w;
                s1 += x0 + x1 + x2 + x3;
                s2 += x0 * x0 + x1 * x1 + x2 * x2 + x3 * x3;
            }
#pragma unroll
            for (int off = 16; off > 0; off >>= 1) {
                s1 += __shfl_xor(s1, off, 32);
                s2 += __shfl_xor(s2, off, 32);
            }
            const float mean = s1 * (1.f / HD);
            const float var  = s2 * (1.f / HD) - mean * mean;
            const float inv  = rsqrtf(var + LN_EPS);
            float* orow = out + ((long)b * Tc + t) * HD;
#pragma unroll
            for (int c0 = lane * 4; c0 < HD; c0 += 128) {
                float4 o4 = *(const float4*)(out_lds + b * HD + c0);
                float4 r4 = *(const float4*)(res + c0);
                float4 g4 = *(const float4*)(ln_g + c0);
                float4 e4 = *(const float4*)(ln_b + c0);
                float4 y;
                y.x = (o4.x + r4.x - mean) * inv * g4.x + e4.x;
                y.y = (o4.y + r4.y - mean) * inv * g4.y + e4.y;
                y.z = (o4.z + r4.z - mean) * inv * g4.z + e4.z;
                y.w = (o4.w + r4.w - mean) * inv * g4.w + e4.w;
                *(float4*)(orow + c0) = y;
                short4 o;
                o.x = f2bf(y.x); o.y = f2bf(y.y); o.z = f2bf(y.z); o.w = f2bf(y.w);
                *(short4*)(abuf + b * HD + c0) = o;
            }
        }
        __syncthreads();
    }
}

extern "C" void kernel_launch(void* const* d_in, const int* in_sizes, int n_in,
                              void* d_out, int out_size, void* d_ws, size_t ws_size,
                              hipStream_t stream) {
    const float* xs = (const float*)d_in[0];
    // d_in[1] = ilens (all == S, unused)
    const float* Wq = (const float*)d_in[2];
    const float* bq = (const float*)d_in[3];
    const float* Wk = (const float*)d_in[4];
    const float* bk = (const float*)d_in[5];
    const float* Wv = (const float*)d_in[6];
    const float* bv = (const float*)d_in[7];
    const float* Wo = (const float*)d_in[8];
    const float* bo = (const float*)d_in[9];
    const float* lg = (const float*)d_in[10];
    const float* lb = (const float*)d_in[11];

    // workspace layout
    float* keys   = (float*)d_ws;                              // [B,S,AD]   16 MB
    float* values = keys + (size_t)Bc * Sc * AD;               // [B,S,HD]   64 MB
    short* xsb    = (short*)(values + (size_t)Bc * Sc * HD);   // bf16 xs    32 MB
    short* Wqp    = xsb + (size_t)Bc * Sc * HD;                // packed Wq  128 KB
    short* Wkp    = Wqp + (size_t)HD * AD;                     // packed Wk  128 KB
    short* Wvp    = Wkp + (size_t)HD * AD;                     // packed Wv  512 KB
    short* Wop    = Wvp + (size_t)HD * HD;                     // packed Wo  512 KB
    float* outp   = (float*)d_out;                             // [B,T,HD]

    const long nxs = (long)Bc * Sc * HD;
    cvt_bf_kernel<<<(int)((nxs / 4 + 255) / 256), 256, 0, stream>>>(xs, xsb, nxs);
    pack_w_kernel<<<(HD * AD + 255) / 256, 256, 0, stream>>>(Wq, Wqp, HD, AD);
    pack_w_kernel<<<(HD * AD + 255) / 256, 256, 0, stream>>>(Wk, Wkp, HD, AD);
    pack_w_kernel<<<(HD * HD + 255) / 256, 256, 0, stream>>>(Wv, Wvp, HD, HD);
    pack_w_kernel<<<(HD * HD + 255) / 256, 256, 0, stream>>>(Wo, Wop, HD, HD);

    kv_kernel<<<(Bc * Sc) / 16, 256, 0, stream>>>(xsb, Wkp, bk, Wvp, bv, keys, values);
    scan_kernel<<<1, 512, 0, stream>>>(xs, Wqp, bq, keys, values, Wop, bo, lg, lb, outp);
}
